// CrossAttention_78735340470832
// MI455X (gfx1250) — matrix operand order repaired and checked by reading, not compiled, not run on hardware
//
#include <hip/hip_runtime.h>
#include <hip/hip_bf16.h>
#include <math.h>

typedef __attribute__((ext_vector_type(16))) _Float16 v16h;
typedef __attribute__((ext_vector_type(8)))  _Float16 v8h;
typedef __attribute__((ext_vector_type(8)))  float    v8f;

namespace ca {
constexpr int B = 8, N = 2048, M = 2048, C = 512, H = 8, DK = 64;
constexpr float SCALE = 0.125f;   // 1/sqrt(DK)
constexpr float EPS_LN = 1e-5f;
}

// ---------------------------------------------------------------------------
// WMMA fragment helpers (wave32, 16-bit operands; contiguous k order used for both A and B)
// A (16x32, MxK): lane<16 -> row=lane, K k0..k0+15
//                 lane>=16 -> row=lane-16, K k0+16..k0+31
// B (32x16, KxN): source stored N-major / K-contiguous: elem(k,n)=base[n*ld+k]
//                 lane<16 -> col=lane, K k0..k0+15 ; lane>=16 -> K k0+16..k0+31
// ---------------------------------------------------------------------------
__device__ inline v16h load_frag_a16(const _Float16* base, int ld, int lane, int k0) {
  const int r  = lane & 15;
  const int kb = k0 + ((lane & 16) ? 16 : 0);
  const _Float16* p = base + (size_t)r * ld + kb;
  v8h lo = *(const v8h*)p;
  v8h hi = *(const v8h*)(p + 8);
  v16h a;
#pragma unroll
  for (int i = 0; i < 8; ++i) { a[i] = lo[i]; a[8 + i] = hi[i]; }
  return a;
}

__device__ inline v16h load_frag_b16(const _Float16* base, int ld, int lane, int k0) {
  const int n  = lane & 15;
  const int kb = k0 + ((lane & 16) ? 16 : 0);
  return *(const v16h*)(base + (size_t)n * ld + kb);
}

__device__ inline v8f wmma_f16(v16h a, v16h b, v8f c) {
  return __builtin_amdgcn_wmma_f32_16x16x32_f16(false, a, false, b, (short)0, c,
                                                false, false);
}

// ---------------------------------------------------------------------------
// Kernel 1: fused LayerNorm for x and y -> f16
// ---------------------------------------------------------------------------
__global__ __launch_bounds__(128) void ln_xy_kernel(
    const float* __restrict__ x, const float* __restrict__ y,
    const float* __restrict__ gx, const float* __restrict__ bx,
    const float* __restrict__ gy, const float* __restrict__ by,
    _Float16* __restrict__ xn, _Float16* __restrict__ yn)
{
  using namespace ca;
  const int row = blockIdx.x;
  const bool isY = row >= B * N;
  const int r2 = isY ? (row - B * N) : row;
  const float* src = (isY ? y : x) + (size_t)r2 * C;
  const float* gg = isY ? gy : gx;
  const float* bb = isY ? by : bx;
  _Float16* dst = (isY ? yn : xn) + (size_t)r2 * C;

  float v[4]; float s = 0.f, sq = 0.f;
#pragma unroll
  for (int i = 0; i < 4; ++i) {
    v[i] = src[threadIdx.x + i * 128];
    s += v[i]; sq += v[i] * v[i];
  }
  __shared__ float s1[128], s2[128];
  s1[threadIdx.x] = s; s2[threadIdx.x] = sq;
  __syncthreads();
  for (int st = 64; st > 0; st >>= 1) {
    if (threadIdx.x < st) {
      s1[threadIdx.x] += s1[threadIdx.x + st];
      s2[threadIdx.x] += s2[threadIdx.x + st];
    }
    __syncthreads();
  }
  const float mean = s1[0] * (1.f / C);
  const float var  = s2[0] * (1.f / C) - mean * mean;
  const float inv  = rsqrtf(var + EPS_LN);
#pragma unroll
  for (int i = 0; i < 4; ++i) {
    int c = threadIdx.x + i * 128;
    dst[c] = (_Float16)(((v[i] - mean) * inv) * gg[c] + bb[c]);
  }
}

// ---------------------------------------------------------------------------
// Kernel 2: weight prep. Wq/Wk/Wv (C x C, [in][out]) -> f16 transposed [out][in]
//           Wc ([out][in]) -> f16 as-is. All become K-contiguous B-frag sources.
// ---------------------------------------------------------------------------
__global__ __launch_bounds__(256) void wprep_kernel(
    const float* __restrict__ Wq, const float* __restrict__ Wk,
    const float* __restrict__ Wv, const float* __restrict__ Wc,
    _Float16* __restrict__ wqT, _Float16* __restrict__ wkT,
    _Float16* __restrict__ wvT, _Float16* __restrict__ wc16)
{
  using namespace ca;
  const int idx = blockIdx.x * 256 + threadIdx.x;       // < 4*C*C = 2^20
  const int w   = idx >> 18;                            // C*C == 2^18
  const int rem = idx & (C * C - 1);
  const int j = rem >> 9, c = rem & 511;                // dest [j][c]
  if (w == 0)      wqT[rem]  = (_Float16)Wq[(size_t)c * C + j];
  else if (w == 1) wkT[rem]  = (_Float16)Wk[(size_t)c * C + j];
  else if (w == 2) wvT[rem]  = (_Float16)Wv[(size_t)c * C + j];
  else             wc16[rem] = (_Float16)Wc[rem];
}

// ---------------------------------------------------------------------------
// Kernel 3: WMMA GEMM projections.  out[row, col] = A[row,:] . WT[col,:] + bias
// Block: 128 rows x 64 cols, 8 waves (4x2), wave = 32x32 (2x2 frags), K-step 32.
// mode 0: Q -> q16 [b,h,n,dk] + qflat fp32 [b,n,c]
// mode 1: K -> k16 [b,h,m,dk]
// mode 2: V -> vT  [b,h,dk,m]   (transposed for P@V B-fragments)
// ---------------------------------------------------------------------------
__global__ __launch_bounds__(256) void proj_gemm_kernel(
    const _Float16* __restrict__ A, const _Float16* __restrict__ WT,
    const float* __restrict__ bias,
    _Float16* __restrict__ out16, float* __restrict__ outf32, int mode)
{
  using namespace ca;
  const int lane = threadIdx.x & 31;
  const int wave = threadIdx.x >> 5;
  const int rows0 = blockIdx.x * 128 + (wave >> 1) * 32;
  const int cols0 = blockIdx.y * 64  + (wave & 1) * 32;

  v8f acc[2][2] = {};
  for (int kk = 0; kk < C; kk += 32) {
    v16h af0 = load_frag_a16(A + (size_t)rows0 * C, C, lane, kk);
    v16h af1 = load_frag_a16(A + (size_t)(rows0 + 16) * C, C, lane, kk);
    v16h bf0 = load_frag_b16(WT + (size_t)cols0 * C, C, lane, kk);
    v16h bf1 = load_frag_b16(WT + (size_t)(cols0 + 16) * C, C, lane, kk);
    acc[0][0] = wmma_f16(af0, bf0, acc[0][0]);
    acc[0][1] = wmma_f16(af0, bf1, acc[0][1]);
    acc[1][0] = wmma_f16(af1, bf0, acc[1][0]);
    acc[1][1] = wmma_f16(af1, bf1, acc[1][1]);
  }

  const int colLane = lane & 15;
  const int rhalf   = (lane & 16) ? 8 : 0;
#pragma unroll
  for (int fc = 0; fc < 2; ++fc) {
    const int col  = cols0 + fc * 16 + colLane;
    const float bv = bias[col];
    const int head = col >> 6, d = col & 63;
#pragma unroll
    for (int fr = 0; fr < 2; ++fr) {
#pragma unroll
      for (int i = 0; i < 8; ++i) {
        const int row  = rows0 + fr * 16 + i + rhalf;   // in [0, B*2048)
        const int bidx = row >> 11;
        const int s    = row & 2047;
        const float v  = acc[fr][fc][i] + bv;
        if (mode == 0) {
          outf32[(size_t)row * C + col] = v;
          out16[(((size_t)bidx * H + head) * N + s) * DK + d] = (_Float16)v;
        } else if (mode == 1) {
          out16[(((size_t)bidx * H + head) * M + s) * DK + d] = (_Float16)v;
        } else {
          out16[(((size_t)bidx * H + head) * DK + d) * M + s] = (_Float16)v;
        }
      }
    }
  }
}

// ---------------------------------------------------------------------------
// Kernel 4: attention pass 1 — per (b,h,n): row max & sum of exp (online).
// Block = (bh, 16-row n tile); 8 waves stripe the 128 m-tiles; LDS combine.
// ---------------------------------------------------------------------------
__global__ __launch_bounds__(256) void attn_stats_kernel(
    const _Float16* __restrict__ q16, const _Float16* __restrict__ k16,
    float* __restrict__ rmax, float* __restrict__ rsum)
{
  using namespace ca;
  const int lane = threadIdx.x & 31;
  const int wave = threadIdx.x >> 5;
  const int bh = blockIdx.x;
  const int n0 = blockIdx.y * 16;

  const _Float16* qbase = q16 + ((size_t)bh * N + n0) * DK;
  const v16h aq0 = load_frag_a16(qbase, DK, lane, 0);
  const v16h aq1 = load_frag_a16(qbase, DK, lane, 32);

  float mx[8], sm[8];
#pragma unroll
  for (int i = 0; i < 8; ++i) { mx[i] = -1e30f; sm[i] = 0.f; }

  for (int mt = wave; mt < M / 16; mt += 8) {
    const _Float16* kp = k16 + ((size_t)bh * M + mt * 16) * DK;
    v16h b0 = load_frag_b16(kp, DK, lane, 0);
    v16h b1 = load_frag_b16(kp, DK, lane, 32);
    v8f c = {};
    c = wmma_f16(aq0, b0, c);
    c = wmma_f16(aq1, b1, c);
#pragma unroll
    for (int i = 0; i < 8; ++i) {
      const float s  = c[i] * SCALE;
      const float nm = fmaxf(mx[i], s);
      sm[i] = sm[i] * __expf(mx[i] - nm) + __expf(s - nm);
      mx[i] = nm;
    }
  }

  __shared__ float smax[8][32][8];
  __shared__ float ssum[8][32][8];
#pragma unroll
  for (int i = 0; i < 8; ++i) { smax[wave][lane][i] = mx[i]; ssum[wave][lane][i] = sm[i]; }
  __syncthreads();

  if (threadIdx.x < 16) {
    const int row = threadIdx.x;
    const int l0 = (row < 8) ? 0 : 16;
    const int vi = row & 7;
    float Mv = -1e30f, Sv = 0.f;
    for (int w = 0; w < 8; ++w)
      for (int l = 0; l < 16; ++l) {
        const float m2 = smax[w][l0 + l][vi];
        const float s2 = ssum[w][l0 + l][vi];
        const float nm = fmaxf(Mv, m2);
        Sv = Sv * __expf(Mv - nm) + s2 * __expf(m2 - nm);
        Mv = nm;
      }
    rmax[(size_t)bh * N + n0 + row] = Mv;
    rsum[(size_t)bh * N + n0 + row] = Sv;
  }
}

// ---------------------------------------------------------------------------
// Kernel 5: attention pass 2 — all 8 heads co-resident (wave == head).
// Per 16-m subtile: p_h = exp(s-mx)/Z_h ; D = 1e-9 + sum_h p_h (via LDS);
// normalized p written to LDS in A-frag layout (16x32), then P@V with WMMA.
// ---------------------------------------------------------------------------
__global__ __launch_bounds__(256) void attn_pv_kernel(
    const _Float16* __restrict__ q16, const _Float16* __restrict__ k16,
    const _Float16* __restrict__ vT,
    const float* __restrict__ rmax, const float* __restrict__ rsum,
    float* __restrict__ oflat)
{
  using namespace ca;
  const int lane = threadIdx.x & 31;
  const int h    = threadIdx.x >> 5;         // wave == head
  const int b    = blockIdx.x;
  const int n0   = blockIdx.y * 16;
  const int bh   = b * H + h;
  const int colLane = lane & 15;
  const int rhalf   = (lane & 16) ? 8 : 0;

  const _Float16* qbase = q16 + ((size_t)bh * N + n0) * DK;
  const v16h aq0 = load_frag_a16(qbase, DK, lane, 0);
  const v16h aq1 = load_frag_a16(qbase, DK, lane, 32);

  float mx[8], sm[8];
#pragma unroll
  for (int i = 0; i < 8; ++i) {
    const int n = n0 + i + rhalf;
    mx[i] = rmax[(size_t)bh * N + n];
    sm[i] = rsum[(size_t)bh * N + n];
  }

  __shared__ float    pf[8][16][16];     // raw softmax probs, per head
  __shared__ _Float16 Pa[8][16 * 32];    // normalized probs, A-frag layout
  v8f oacc[4] = {};

  for (int m0 = 0; m0 < M; m0 += 32) {
    float pv[2][8];
#pragma unroll
    for (int sub = 0; sub < 2; ++sub) {
      const int mt = m0 + sub * 16;
      const _Float16* kp = k16 + ((size_t)bh * M + mt) * DK;
      v16h b0 = load_frag_b16(kp, DK, lane, 0);
      v16h b1 = load_frag_b16(kp, DK, lane, 32);
      v8f c = {};
      c = wmma_f16(aq0, b0, c);
      c = wmma_f16(aq1, b1, c);
#pragma unroll
      for (int i = 0; i < 8; ++i) {
        const float p = __expf(c[i] * SCALE - mx[i]) / sm[i];
        pv[sub][i] = p;
        pf[h][i + rhalf][colLane] = p;
      }
      __syncthreads();
#pragma unroll
      for (int i = 0; i < 8; ++i) {
        const int r = i + rhalf;
        float D = 1e-9f;
#pragma unroll
        for (int hh = 0; hh < 8; ++hh) D += pf[hh][r][colLane];
        Pa[h][r * 32 + sub * 16 + colLane] = (_Float16)(pv[sub][i] / D);
      }
      __syncthreads();
    }
    const v16h aP = load_frag_a16(&Pa[h][0], 32, lane, 0);
#pragma unroll
    for (int dc = 0; dc < 4; ++dc) {
      v16h bv = load_frag_b16(vT + ((size_t)bh * DK + dc * 16) * M + m0, M, lane, 0);
      oacc[dc] = wmma_f16(aP, bv, oacc[dc]);
    }
  }

#pragma unroll
  for (int dc = 0; dc < 4; ++dc)
#pragma unroll
    for (int i = 0; i < 8; ++i) {
      const int n   = n0 + i + rhalf;
      const int col = h * DK + dc * 16 + colLane;
      oflat[((size_t)b * N + n) * C + col] = oacc[dc][i];
    }
}

// ---------------------------------------------------------------------------
// Kernel 6: h = (q - o) @ Wc^T + bc ; LN ; leaky-relu ; out = q + h
// Block = 16 rows x full 512 cols; diff staged in LDS as f16 A-tile.
// ---------------------------------------------------------------------------
__global__ __launch_bounds__(256) void out_kernel(
    const float* __restrict__ qflat, const float* __restrict__ oflat,
    const _Float16* __restrict__ wc16, const float* __restrict__ bc,
    const float* __restrict__ gO, const float* __restrict__ bO,
    float* __restrict__ out)
{
  using namespace ca;
  const int g0 = blockIdx.x * 16;
  __shared__ _Float16 dT[16 * 512];
  __shared__ float ssum[16], ssq[16];

  for (int idx = threadIdx.x; idx < 16 * C; idx += 256) {
    const int r = idx >> 9, c = idx & 511;
    const size_t g = (size_t)(g0 + r) * C + c;
    dT[idx] = (_Float16)(qflat[g] - oflat[g]);
  }
  if (threadIdx.x < 16) { ssum[threadIdx.x] = 0.f; ssq[threadIdx.x] = 0.f; }
  __syncthreads();

  const int lane = threadIdx.x & 31;
  const int wave = threadIdx.x >> 5;
  const int colBase = wave * 64;
  const int colLane = lane & 15;
  const int rhalf   = (lane & 16) ? 8 : 0;

  v8f acc[4] = {};
  for (int kk = 0; kk < C; kk += 32) {
    const v16h a = load_frag_a16(dT, C, lane, kk);
#pragma unroll
    for (int fc = 0; fc < 4; ++fc) {
      v16h bW = load_frag_b16(wc16 + (size_t)(colBase + fc * 16) * C, C, lane, kk);
      acc[fc] = wmma_f16(a, bW, acc[fc]);
    }
  }

  float hv[4][8];
#pragma unroll
  for (int fc = 0; fc < 4; ++fc) {
    const float bcv = bc[colBase + fc * 16 + colLane];
#pragma unroll
    for (int i = 0; i < 8; ++i) hv[fc][i] = acc[fc][i] + bcv;
  }
#pragma unroll
  for (int i = 0; i < 8; ++i) {
    const int r = i + rhalf;
    float ps = 0.f, pq = 0.f;
#pragma unroll
    for (int fc = 0; fc < 4; ++fc) { const float v = hv[fc][i]; ps += v; pq += v * v; }
    atomicAdd(&ssum[r], ps);
    atomicAdd(&ssq[r], pq);
  }
  __syncthreads();

#pragma unroll
  for (int fc = 0; fc < 4; ++fc) {
    const int col = colBase + fc * 16 + colLane;
    const float g = gO[col], bb = bO[col];
#pragma unroll
    for (int i = 0; i < 8; ++i) {
      const int r = i + rhalf;
      const float mean = ssum[r] * (1.f / C);
      const float var  = ssq[r] * (1.f / C) - mean * mean;
      const float inv  = rsqrtf(var + EPS_LN);
      const float ln   = (hv[fc][i] - mean) * inv * g + bb;
      const float lr   = ln > 0.f ? ln : 0.02f * ln;
      const size_t gi  = (size_t)(g0 + r) * C + col;
      out[gi] = qflat[gi] + lr;
    }
  }
}

// ---------------------------------------------------------------------------
extern "C" void kernel_launch(void* const* d_in, const int* in_sizes, int n_in,
                              void* d_out, int out_size, void* d_ws, size_t ws_size,
                              hipStream_t stream) {
  using namespace ca;
  const float* x  = (const float*)d_in[0];
  const float* y  = (const float*)d_in[1];
  const float* gx = (const float*)d_in[2];
  const float* bx = (const float*)d_in[3];
  const float* gy = (const float*)d_in[4];
  const float* by = (const float*)d_in[5];
  const float* Wq = (const float*)d_in[6];
  const float* bq = (const float*)d_in[7];
  const float* Wk = (const float*)d_in[8];
  const float* bk = (const float*)d_in[9];
  const float* Wv = (const float*)d_in[10];
  const float* bv = (const float*)d_in[11];
  const float* Wc = (const float*)d_in[12];
  const float* bc = (const float*)d_in[13];
  const float* gO = (const float*)d_in[14];
  const float* bO = (const float*)d_in[15];
  float* out = (float*)d_out;

  char* basep = (char*)d_ws;
  size_t off = 0;
  auto take = [&](size_t bytes) -> void* {
    void* p = basep + off;
    off += (bytes + 255) & ~(size_t)255;
    return p;
  };
  _Float16* xn   = (_Float16*)take((size_t)B * N * C * 2);
  _Float16* yn   = (_Float16*)take((size_t)B * M * C * 2);
  _Float16* wqT  = (_Float16*)take((size_t)C * C * 2);
  _Float16* wkT  = (_Float16*)take((size_t)C * C * 2);
  _Float16* wvT  = (_Float16*)take((size_t)C * C * 2);
  _Float16* wc16 = (_Float16*)take((size_t)C * C * 2);
  _Float16* q16  = (_Float16*)take((size_t)B * H * N * DK * 2);
  _Float16* k16  = (_Float16*)take((size_t)B * H * M * DK * 2);
  _Float16* vTt  = (_Float16*)take((size_t)B * H * M * DK * 2);
  float* qflat = (float*)take((size_t)B * N * C * 4);
  float* oflat = (float*)take((size_t)B * N * C * 4);
  float* rmax  = (float*)take((size_t)B * H * N * 4);
  float* rsum  = (float*)take((size_t)B * H * N * 4);

  ln_xy_kernel<<<B * (N + M), 128, 0, stream>>>(x, y, gx, bx, gy, by, xn, yn);
  wprep_kernel<<<(4 * C * C) / 256, 256, 0, stream>>>(Wq, Wk, Wv, Wc, wqT, wkT, wvT, wc16);

  dim3 gp(B * N / 128, C / 64);
  proj_gemm_kernel<<<gp, 256, 0, stream>>>(xn, wqT, bq, q16, qflat, 0);
  proj_gemm_kernel<<<gp, 256, 0, stream>>>(yn, wkT, bk, k16, nullptr, 1);
  proj_gemm_kernel<<<gp, 256, 0, stream>>>(yn, wvT, bv, vTt, nullptr, 2);

  attn_stats_kernel<<<dim3(B * H, N / 16), 256, 0, stream>>>(q16, k16, rmax, rsum);
  attn_pv_kernel<<<dim3(B, N / 16), 256, 0, stream>>>(q16, k16, vTt, rmax, rsum, oflat);
  out_kernel<<<B * N / 16, 256, 0, stream>>>(qflat, oflat, wc16, bc, gO, bO, out);

  (void)in_sizes; (void)n_in; (void)out_size; (void)ws_size;
}
